// MoE_7842610283167
// MI455X (gfx1250) — compile-verified, matrix operand order checked by reading, not run on hardware
//
#include <hip/hip_runtime.h>
#include <hip/hip_bf16.h>
#include <math.h>

// ---------------------------------------------------------------------------
// MoE (shared expert + top-3 of 31 routed experts) for gfx1250 / MI455X.
// Sparse routing + fused per-tile double-GEMM in 320KB LDS, f16 WMMA with
// f32 accumulation (v_wmma_f32_16x16x32_f16).
// W tiles are staged into LDS pre-swizzled into the WMMA B-fragment layout
// so every B fragment is one contiguous 32B read (2x ds_load_b128).
// ---------------------------------------------------------------------------

typedef _Float16 h16;
typedef __attribute__((ext_vector_type(16))) _Float16 v16h;
typedef __attribute__((ext_vector_type(8)))  _Float16 v8h;
typedef __attribute__((ext_vector_type(8)))  float    v8f;

#define NTOK   8192          // B*S
#define DIM    1024
#define INTER  1024
#define NEXP   31
#define TOPK   3
#define MT     64            // token rows per workgroup tile
#define NT     64            // output cols per chunk
#define KC     32            // K step (matches wmma 16x16x32)
#define WPELEM 2048          // packed W tile: 128 fragment-lanes * 16 halfs

// Branchless tanh-form GELU (v_exp_f32 trans op; no OCML branches)
__device__ __forceinline__ float gelu_fast(float v) {
    float u = v * (0.7978845608028654f + 0.035677408136300125f * v * v);
    u = fminf(fmaxf(u, -9.f), 9.f);
    const float e2 = __expf(2.f * u);
    const float th = (e2 - 1.f) / (e2 + 1.f);     // tanh(u)
    return 0.5f * v * (1.f + th);
}

// ---- WMMA fragment loaders (ISA 7.12.2 layouts, wave32) --------------------
// A 16x32 f16 from row-major LDS: lane&15 = row M; lane>=16 -> K base +8.
// elems 0..7 -> K = kb+0..7 ; elems 8..15 -> K = kb+16..23  (two 16B chunks)
__device__ __forceinline__ v16h frag_a(const h16* __restrict__ base, int ld,
                                       int row0, int k0, int lane) {
    const h16* p = base + (row0 + (lane & 15)) * ld + k0 + ((lane & 16) ? 8 : 0);
    v16h a;
#pragma unroll
    for (int e = 0; e < 8; ++e)  a[e] = p[e];
#pragma unroll
    for (int e = 8; e < 16; ++e) a[e] = p[e + 8];
    return a;
}

// B fragment from the packed layout: fragment-lane f = ct*32 + lane owns 16
// contiguous halfs (32B).  ct = 16-column sub-tile index (0..3).
__device__ __forceinline__ v16h frag_b_packed(const h16* __restrict__ Wp,
                                              int ct, int lane) {
    return *(const v16h*)(Wp + (((ct << 5) + lane) << 4));
}

// Stage a KC x NT fp32 weight tile into LDS, swizzled into B-fragment order.
// thread t: fragment-lane f = t&127, half-part p = t>>7.
//   col = (f>>5)*16 + (f&15);  kb = (f&16)?8:0;  K = kb + p*16 + i, e = p*8+i
// -> 8 strided global_load_b32 + one 16B contiguous ds_store_b128.
__device__ __forceinline__ void stage_w_packed(const float* __restrict__ Wg, int ld,
                                               int k0, int n0,
                                               h16* __restrict__ Wp) {
    const int t   = threadIdx.x;
    const int f   = t & 127;
    const int p   = t >> 7;
    const int col = ((f >> 5) << 4) + (f & 15);
    const int kb  = (f & 16) ? 8 : 0;
    const float* g = Wg + (size_t)(k0 + kb + p * 16) * ld + n0 + col;
    v8h val;
#pragma unroll
    for (int i = 0; i < 8; ++i) val[i] = (h16)g[(size_t)i * ld];
    *(v8h*)(Wp + f * 16 + p * 8) = val;
    // prefetch the K+2 tile this thread will stage next (global_prefetch_b8)
    __builtin_prefetch(g + (size_t)(2 * KC) * ld, 0, 1);
}

// ---------------------------------------------------------------------------
__global__ void moe_init_counts(int* counts) {
    if (threadIdx.x < NEXP) counts[threadIdx.x] = 0;
}

// Router: one wave per token. lane e (<31) owns expert e's logit.
__global__ void moe_router(const float* __restrict__ x,
                           const float* __restrict__ Wr,
                           const float* __restrict__ br,
                           int* __restrict__ counts,
                           int* __restrict__ lists,
                           float* __restrict__ glists) {
    const int lane = threadIdx.x & 31;
    const int wave = threadIdx.x >> 5;
    const int tok  = blockIdx.x * 8 + wave;
    const float* xr = x + (size_t)tok * DIM;

    float xreg[32];
#pragma unroll
    for (int i = 0; i < 32; ++i) xreg[i] = xr[lane + 32 * i];

    float mylogit = -INFINITY;
    for (int e = 0; e < NEXP; ++e) {
        float p = 0.f;
#pragma unroll
        for (int i = 0; i < 32; ++i) p += xreg[i] * Wr[(lane + 32 * i) * NEXP + e];
#pragma unroll
        for (int off = 16; off > 0; off >>= 1) p += __shfl_xor(p, off, 32);
        if (lane == e) mylogit = p + br[e];
    }
    // softmax across lanes (lane 31 invalid)
    float m = mylogit;
#pragma unroll
    for (int off = 16; off > 0; off >>= 1) m = fmaxf(m, __shfl_xor(m, off, 32));
    float ex = (lane < NEXP) ? __expf(mylogit - m) : 0.f;
    float s = ex;
#pragma unroll
    for (int off = 16; off > 0; off >>= 1) s += __shfl_xor(s, off, 32);
    const float prob = ex / s;

    // top-3 by iterated warp max; winner lane appends to its expert bucket
    float v = (lane < NEXP) ? prob : -1.f;
    for (int k = 0; k < TOPK; ++k) {
        float mx = v;
#pragma unroll
        for (int off = 16; off > 0; off >>= 1) mx = fmaxf(mx, __shfl_xor(mx, off, 32));
        int cand = (v == mx) ? lane : 32;
#pragma unroll
        for (int off = 16; off > 0; off >>= 1) cand = min(cand, __shfl_xor(cand, off, 32));
        if (lane == cand) {
            const int pos = atomicAdd(&counts[lane], 1);
            lists[lane * NTOK + pos]  = tok;
            glists[lane * NTOK + pos] = mx;
            v = -1.f;   // remove from next round
        }
    }
}

// ---------------------------------------------------------------------------
// Core fused double-GEMM body shared by both FFN kernels.
//   Xt: MT x DIM f16 (LDS)   Ht: MT x INTER f16 (LDS)
//   Wp: 2 x WPELEM f16 (LDS, double-buffered packed W tiles)
// Accumulators live in per-wave v8f pairs; one barrier per K step.

// GEMM: acc over K, A from LDS (row-major, ld), W from global via packed tiles
#define FUSED_GEMM_LOOP(A_BASE, A_LD, W_GLOBAL, W_LD, KDIM, N0)                          \
    v8f acc0 = {}; v8f acc1 = {};                                                        \
    stage_w_packed(W_GLOBAL, W_LD, 0, (N0), Wp);                                         \
    for (int k0 = 0; k0 < (KDIM); k0 += KC) {                                            \
        __syncthreads();                                                                 \
        h16* cur = Wp + ((k0 & KC) ? WPELEM : 0);                                        \
        h16* nxt = Wp + ((k0 & KC) ? 0 : WPELEM);                                        \
        if (k0 + KC < (KDIM)) stage_w_packed(W_GLOBAL, W_LD, k0 + KC, (N0), nxt);        \
        const v16h a  = frag_a((A_BASE), (A_LD), wm * 16, k0, lane);                     \
        const v16h b0 = frag_b_packed(cur, wn * 2, lane);                                \
        const v16h b1 = frag_b_packed(cur, wn * 2 + 1, lane);                            \
        acc0 = __builtin_amdgcn_wmma_f32_16x16x32_f16(false, a, false, b0, (short)0,     \
                                                      acc0, false, false);               \
        acc1 = __builtin_amdgcn_wmma_f32_16x16x32_f16(false, a, false, b1, (short)0,     \
                                                      acc1, false, false);               \
    }

// Shared expert: out = x + gelu(x@W1+b1)@W2 + b2   (fused in one workgroup)
__global__ void moe_shared_ffn(const float* __restrict__ x,
                               const float* __restrict__ W1, const float* __restrict__ b1,
                               const float* __restrict__ W2, const float* __restrict__ b2,
                               float* __restrict__ out) {
    extern __shared__ char smem[];
    h16* Xt = (h16*)smem;              // MT x DIM   (128KB)
    h16* Ht = Xt + MT * DIM;           // MT x INTER (128KB)
    h16* Wp = Ht + MT * INTER;         // 2 x WPELEM (8KB)

    const int r0   = blockIdx.x * MT;
    const int t    = threadIdx.x;
    const int lane = t & 31;
    const int wave = t >> 5;
    const int wm   = wave & 3;         // 4 M sub-tiles (16 rows each)
    const int wn   = wave >> 2;        // 2 N sub-tiles (each wave: 16 x 32)

    // X tile: fp32 -> f16 into LDS
    for (int it = 0; it < 64; ++it) {
        const int idx = t + it * 256;            // 16384 float4s
        const int r   = idx >> 8;
        const int c4  = (idx & 255) << 2;
        const float4 fv = *(const float4*)(x + (size_t)(r0 + r) * DIM + c4);
        h16* d = Xt + r * DIM + c4;
        d[0] = (h16)fv.x; d[1] = (h16)fv.y; d[2] = (h16)fv.z; d[3] = (h16)fv.w;
    }
    __syncthreads();

    // ---- GEMM1 + bias + GELU -> Ht ----
    for (int n0 = 0; n0 < INTER; n0 += NT) {
        FUSED_GEMM_LOOP(Xt, DIM, W1, INTER, DIM, n0)
        const int rb = wm * 16 + ((lane & 16) ? 8 : 0);
        const int cA = n0 + wn * 32 + (lane & 15);
        const int cB = cA + 16;
#pragma unroll
        for (int j = 0; j < 8; ++j) {
            const int row = rb + j;
            Ht[row * INTER + cA] = (h16)gelu_fast(acc0[j] + b1[cA]);
            Ht[row * INTER + cB] = (h16)gelu_fast(acc1[j] + b1[cB]);
        }
    }
    __syncthreads();

    // ---- GEMM2 -> out = x + (Ht @ W2 + b2) ----
    for (int n0 = 0; n0 < DIM; n0 += NT) {
        FUSED_GEMM_LOOP(Ht, INTER, W2, DIM, INTER, n0)
        const int rb = wm * 16 + ((lane & 16) ? 8 : 0);
        const int cA = n0 + wn * 32 + (lane & 15);
        const int cB = cA + 16;
#pragma unroll
        for (int j = 0; j < 8; ++j) {
            const size_t o = (size_t)(r0 + rb + j) * DIM;
            out[o + cA] = x[o + cA] + acc0[j] + b2[cA];
            out[o + cB] = x[o + cB] + acc1[j] + b2[cB];
        }
    }
}

// Routed experts: gathered token tiles per expert, atomic scatter-add with gates
__global__ void moe_routed_ffn(const float* __restrict__ x,
                               const float* __restrict__ We1, const float* __restrict__ Be1,
                               const float* __restrict__ We2, const float* __restrict__ Be2,
                               const int* __restrict__ counts,
                               const int* __restrict__ lists,
                               const float* __restrict__ glists,
                               float* __restrict__ out) {
    const int e    = blockIdx.y;
    const int cnt  = counts[e];
    const int tile = blockIdx.x;
    if (tile * MT >= cnt) return;

    extern __shared__ char smem[];
    h16*   Xt   = (h16*)smem;
    h16*   Ht   = Xt + MT * DIM;
    h16*   Wp   = Ht + MT * INTER;
    int*   toks = (int*)(Wp + 2 * WPELEM);
    float* gts  = (float*)(toks + MT);

    const float* W1 = We1 + (size_t)e * DIM * INTER;
    const float* b1 = Be1 + (size_t)e * INTER;
    const float* W2 = We2 + (size_t)e * INTER * DIM;
    const float* b2 = Be2 + (size_t)e * DIM;

    const int t    = threadIdx.x;
    const int lane = t & 31;
    const int wave = t >> 5;
    const int wm   = wave & 3;
    const int wn   = wave >> 2;

    if (t < MT) {
        const int j = tile * MT + t;
        if (j < cnt) { toks[t] = lists[e * NTOK + j]; gts[t] = glists[e * NTOK + j]; }
        else         { toks[t] = -1;                  gts[t] = 0.f; }
    }
    __syncthreads();

    // gather X tile (rows beyond count -> zeros)
    for (int it = 0; it < 64; ++it) {
        const int idx = t + it * 256;
        const int r   = idx >> 8;
        const int c4  = (idx & 255) << 2;
        const int tok = toks[r];
        h16* d = Xt + r * DIM + c4;
        if (tok >= 0) {
            const float4 fv = *(const float4*)(x + (size_t)tok * DIM + c4);
            d[0] = (h16)fv.x; d[1] = (h16)fv.y; d[2] = (h16)fv.z; d[3] = (h16)fv.w;
        } else {
            d[0] = (h16)0.f; d[1] = (h16)0.f; d[2] = (h16)0.f; d[3] = (h16)0.f;
        }
    }
    __syncthreads();

    // ---- GEMM1 + bias + GELU -> Ht ----
    for (int n0 = 0; n0 < INTER; n0 += NT) {
        FUSED_GEMM_LOOP(Xt, DIM, W1, INTER, DIM, n0)
        const int rb = wm * 16 + ((lane & 16) ? 8 : 0);
        const int cA = n0 + wn * 32 + (lane & 15);
        const int cB = cA + 16;
#pragma unroll
        for (int j = 0; j < 8; ++j) {
            const int row = rb + j;
            Ht[row * INTER + cA] = (h16)gelu_fast(acc0[j] + b1[cA]);
            Ht[row * INTER + cB] = (h16)gelu_fast(acc1[j] + b1[cB]);
        }
    }
    __syncthreads();

    // ---- GEMM2 -> atomic out += gate * (h @ W2 + b2) ----
    for (int n0 = 0; n0 < DIM; n0 += NT) {
        FUSED_GEMM_LOOP(Ht, INTER, W2, DIM, INTER, n0)
        const int rb = wm * 16 + ((lane & 16) ? 8 : 0);
        const int cA = n0 + wn * 32 + (lane & 15);
        const int cB = cA + 16;
#pragma unroll
        for (int j = 0; j < 8; ++j) {
            const int row = rb + j;
            const int tok = toks[row];
            if (tok >= 0) {
                const float g = gts[row];
                atomicAdd(&out[(size_t)tok * DIM + cA], g * (acc0[j] + b2[cA]));
                atomicAdd(&out[(size_t)tok * DIM + cB], g * (acc1[j] + b2[cB]));
            }
        }
    }
}

// ---------------------------------------------------------------------------
extern "C" void kernel_launch(void* const* d_in, const int* in_sizes, int n_in,
                              void* d_out, int out_size, void* d_ws, size_t ws_size,
                              hipStream_t stream) {
    (void)in_sizes; (void)n_in; (void)out_size; (void)ws_size;
    const float* x   = (const float*)d_in[0];
    const float* ws1 = (const float*)d_in[1];
    const float* bs1 = (const float*)d_in[2];
    const float* ws2 = (const float*)d_in[3];
    const float* bs2 = (const float*)d_in[4];
    const float* We1 = (const float*)d_in[5];
    const float* Be1 = (const float*)d_in[6];
    const float* We2 = (const float*)d_in[7];
    const float* Be2 = (const float*)d_in[8];
    const float* Wr  = (const float*)d_in[9];
    const float* br  = (const float*)d_in[10];
    float* out = (float*)d_out;

    // workspace layout: counts[31] (pad 128B) | lists[31*8192] | gates[31*8192]
    char*  wsb    = (char*)d_ws;
    int*   counts = (int*)wsb;
    int*   lists  = (int*)(wsb + 128);
    float* glists = (float*)(wsb + 128 + (size_t)NEXP * NTOK * sizeof(int));

    const size_t lds1 = (size_t)(MT * DIM + MT * INTER + 2 * WPELEM) * sizeof(h16);
    const size_t lds2 = lds1 + (size_t)MT * (sizeof(int) + sizeof(float));

    // allow >64KB dynamic LDS (CDNA5 WGP: 320KB) — capture-safe host call
    hipFuncSetAttribute((const void*)moe_shared_ffn,
                        hipFuncAttributeMaxDynamicSharedMemorySize, (int)lds1);
    hipFuncSetAttribute((const void*)moe_routed_ffn,
                        hipFuncAttributeMaxDynamicSharedMemorySize, (int)lds2);

    moe_init_counts<<<1, 64, 0, stream>>>(counts);
    moe_router<<<NTOK / 8, 256, 0, stream>>>(x, Wr, br, counts, lists, glists);
    moe_shared_ffn<<<NTOK / MT, 256, lds1, stream>>>(x, ws1, bs1, ws2, bs2, out);
    moe_routed_ffn<<<dim3(NTOK / MT, NEXP), 256, lds2, stream>>>(
        x, We1, Be1, We2, Be2, counts, lists, glists, out);
}
